// MultiScaleRetention_71262097375679
// MI455X (gfx1250) — matrix operand order repaired and checked by reading, not compiled, not run on hardware
//
#include <hip/hip_runtime.h>
#include <hip/hip_bf16.h>

#define B_  4
#define C_  2048
#define E_  1024
#define H_  8
#define HS_ 128
#define BC_ (B_*C_)

typedef __attribute__((ext_vector_type(16))) __bf16 v16bf;
typedef __attribute__((ext_vector_type(4)))  __bf16 v4bf;
typedef __attribute__((ext_vector_type(8)))  float  v8f;

// ---------------- helpers ----------------

__device__ __forceinline__ __bf16 f2bf(float f) {
  unsigned u = __builtin_bit_cast(unsigned, f);
  u += 0x7fffu + ((u >> 16) & 1u);              // round-to-nearest-even
  unsigned short h = (unsigned short)(u >> 16);
  return __builtin_bit_cast(__bf16, h);
}

// kappas[h] = 1 - exp(linspace(log(1/32), log(1/512), 8))[h]; return log(kappa)
__device__ __forceinline__ float lkappa(int h) {
  const float l0 = -3.46573590280f;   // log(1/32)
  const float l1 = -6.23832462504f;   // log(1/512)
  float lam = __expf(l0 + (l1 - l0) * ((float)h * (1.0f / 7.0f)));
  return __logf(1.0f - lam);
}

// A fragment (16x32 bf16): lane m = lane&15; lanes 0-15 hold
// K=0..15; lanes 16-31 hold K=16..31 (matches B fragment order).
__device__ __forceinline__ v16bf load_a_bf16(const __bf16* A, int lda,
                                             int row0, int k0, int lane) {
  int m  = row0 + (lane & 15);
  int kb = (lane & 16) ? 16 : 0;
  const __bf16* p = A + (size_t)m * lda + k0 + kb;
  v16bf a;
#pragma unroll
  for (int j = 0; j < 16; ++j) a[j] = p[j];
  return a;
}

// B fragment (32x16 bf16) from a TRANSPOSED matrix Bt[n][k] (contiguous k):
// lane n = lane&15; lanes 0-15 hold K=0..15, lanes 16-31 hold K=16..31.
__device__ __forceinline__ v16bf load_b_bf16(const __bf16* Bt, int ldb,
                                             int n0, int k0, int lane) {
  int n  = n0 + (lane & 15);
  int kb = (lane & 16) ? 16 : 0;
  const __bf16* p = Bt + (size_t)n * ldb + k0 + kb;
  v16bf b;
#pragma unroll
  for (int j = 0; j < 16; ++j) b[j] = p[j];
  return b;
}

__device__ __forceinline__ v8f wmma_bf16(v16bf a, v16bf b, v8f c) {
  // (neg_a, A, neg_b, B, c_mod, C, reuse_a, reuse_b)
  return __builtin_amdgcn_wmma_f32_16x16x32_bf16(false, a, false, b,
                                                 (short)0, c, false, false);
}

// ---------------- input / weight / state conversion ----------------

// fp32 -> bf16 elementwise (x4 vectorized)
__global__ void k_cvt_in(const float* __restrict__ x, __bf16* __restrict__ y,
                         int n4) {
  int i = blockIdx.x * blockDim.x + threadIdx.x;
  if (i >= n4) return;
  float4 v = ((const float4*)x)[i];
  v4bf o;
  o[0] = f2bf(v.x); o[1] = f2bf(v.y); o[2] = f2bf(v.z); o[3] = f2bf(v.w);
  ((v4bf*)y)[i] = o;
}

// w_q/w_k/w_v (H,E,HS) fp32 -> wt bf16 transposed: wt[n=h*HS+d][e]
__global__ void k_cvt_wqkv(const float* __restrict__ w, __bf16* __restrict__ wt) {
  int idx = blockIdx.x * blockDim.x + threadIdx.x;
  if (idx >= E_ * E_) return;
  int n = idx / E_, e = idx % E_;
  int h = n >> 7, d = n & (HS_ - 1);
  wt[idx] = f2bf(w[((size_t)h * E_ + e) * HS_ + d]);
}

// w (E,E) fp32 -> wt bf16 transposed: wt[n][e] = w[e][n]
__global__ void k_cvt_sq(const float* __restrict__ w, __bf16* __restrict__ wt) {
  int idx = blockIdx.x * blockDim.x + threadIdx.x;
  if (idx >= E_ * E_) return;
  int n = idx / E_, e = idx % E_;
  wt[idx] = f2bf(w[(size_t)e * E_ + n]);
}

// hstate (B,H,HS,HS) fp32 -> ht bf16 transposed within each head: ht[bh][e][d]
__global__ void k_cvt_h(const float* __restrict__ hs, __bf16* __restrict__ ht) {
  int idx = blockIdx.x * blockDim.x + threadIdx.x;
  if (idx >= B_ * H_ * HS_ * HS_) return;
  int bh = idx / (HS_ * HS_);
  int rem = idx % (HS_ * HS_);
  int e = rem / HS_, d = rem % HS_;
  ht[idx] = f2bf(hs[(size_t)bh * HS_ * HS_ + (size_t)d * HS_ + e]);
}

// ---------------- projections q/k/v ----------------
// X (BC,E) bf16 @ Wt (bf16, transposed) -> per-head bf16 layouts.
// mode 0: q -> out0[bh][c][d]
// mode 1: k -> out0[bh][c][d], out1[bh][d][c] * kappa^(C-1-c)  (for state upd)
// mode 2: v -> out0[bh][d][c]
__global__ __launch_bounds__(128) void k_proj(const __bf16* __restrict__ X,
                                              const __bf16* __restrict__ Wt,
                                              __bf16* __restrict__ out0,
                                              __bf16* __restrict__ out1,
                                              int mode) {
  int wv = threadIdx.x >> 5, ln = threadIdx.x & 31;
  int row0 = blockIdx.x * 64 + wv * 16;
  int h = blockIdx.y;               // one 128-col tile == one head
  int n0 = h * HS_;
  v8f z = {}; v8f acc[8];
#pragma unroll
  for (int t = 0; t < 8; ++t) acc[t] = z;
  for (int ks = 0; ks < E_ / 32; ++ks) {
    if (ks + 1 < E_ / 32)
      __builtin_prefetch(X + (size_t)(row0 + (ln & 15)) * E_ + (ks + 1) * 32, 0, 1);
    v16bf a = load_a_bf16(X, E_, row0, ks * 32, ln);
#pragma unroll
    for (int t = 0; t < 8; ++t) {
      v16bf b = load_b_bf16(Wt, E_, n0 + t * 16, ks * 32, ln);
      acc[t] = wmma_bf16(a, b, acc[t]);
    }
  }
  int ln16 = ln & 15, half8 = (ln & 16) ? 8 : 0;
  float lk = lkappa(h);
#pragma unroll
  for (int t = 0; t < 8; ++t) {
    int d = t * 16 + ln16;
#pragma unroll
    for (int r = 0; r < 8; ++r) {
      int rg = row0 + r + half8;
      int bb = rg >> 11, c = rg & (C_ - 1);
      size_t bh = (size_t)bb * H_ + h;
      float val = acc[t][r];
      if (mode == 0) {
        out0[(bh * C_ + c) * HS_ + d] = f2bf(val);
      } else if (mode == 1) {
        out0[(bh * C_ + c) * HS_ + d] = f2bf(val);
        float last = __expf(lk * (float)(C_ - 1 - c));
        out1[(bh * HS_ + d) * C_ + c] = f2bf(val * last);
      } else {
        out0[(bh * HS_ + d) * C_ + c] = f2bf(val);
      }
    }
  }
}

// ---------------- retention (scores * decay) @ v + q @ hstate ----------------
__global__ __launch_bounds__(128) void k_attn(const __bf16* __restrict__ qb,
                                              const __bf16* __restrict__ kb,
                                              const __bf16* __restrict__ vbT,
                                              const __bf16* __restrict__ hT,
                                              float* __restrict__ ret) {
  __shared__ __bf16 sl[4][16][64];   // per-wave bf16 score stripe
  int wv = threadIdx.x >> 5, ln = threadIdx.x & 31;
  int rt = blockIdx.x, h = blockIdx.y, bb = blockIdx.z;
  size_t bh = (size_t)bb * H_ + h;
  const __bf16* q_ = qb  + bh * C_ * HS_;
  const __bf16* k_ = kb  + bh * C_ * HS_;
  const __bf16* v_ = vbT + bh * HS_ * C_;
  const __bf16* h_ = hT  + bh * HS_ * HS_;
  float lk = lkappa(h);
  int row0 = rt * 64 + wv * 16;
  int ln16 = ln & 15, half8 = (ln & 16) ? 8 : 0;
  v8f z = {}; v8f acc[8];
#pragma unroll
  for (int t = 0; t < 8; ++t) acc[t] = z;

  // q A-fragments are invariant across the whole key-tile loop: hoist.
  v16bf qa[4];
#pragma unroll
  for (int ks = 0; ks < 4; ++ks) qa[ks] = load_a_bf16(q_, HS_, row0, ks * 32, ln);

  for (int kt = 0; kt <= rt; ++kt) {
    // S stripe (16 rows x 64 key cols): S = q . k^T, then decay + bf16 -> LDS
#pragma unroll
    for (int jt = 0; jt < 4; ++jt) {
      v8f s = z;
#pragma unroll
      for (int ks = 0; ks < 4; ++ks) {
        v16bf b = load_b_bf16(k_, HS_, kt * 64 + jt * 16, ks * 32, ln);
        s = wmma_bf16(qa[ks], b, s);
      }
      int j = kt * 64 + jt * 16 + ln16;
#pragma unroll
      for (int r = 0; r < 8; ++r) {
        int c = row0 + r + half8;
        int diff = c - j;
        float dec = (diff >= 0) ? __expf(lk * (float)diff) : 0.0f;
        sl[wv][r + half8][jt * 16 + ln16] = f2bf(s[r] * dec);
      }
    }
    __syncthreads();
    // inner += S @ v  (contract 64 key positions, 2 WMMA K-steps)
    v16bf sa[2];
#pragma unroll
    for (int ks2 = 0; ks2 < 2; ++ks2)
      sa[ks2] = load_a_bf16(&sl[wv][0][0], 64, 0, ks2 * 32, ln);
#pragma unroll
    for (int dt = 0; dt < 8; ++dt) {
#pragma unroll
      for (int ks2 = 0; ks2 < 2; ++ks2) {
        v16bf b = load_b_bf16(v_, C_, dt * 16, kt * 64 + ks2 * 32, ln);
        acc[dt] = wmma_bf16(sa[ks2], b, acc[dt]);
      }
    }
    __syncthreads();
  }

  // cross = q @ hstate (contract HS=128)
#pragma unroll
  for (int dt = 0; dt < 8; ++dt) {
#pragma unroll
    for (int ks = 0; ks < 4; ++ks) {
      v16bf b = load_b_bf16(h_, HS_, dt * 16, ks * 32, ln);
      acc[dt] = wmma_bf16(qa[ks], b, acc[dt]);
    }
  }

  // ret[b, c, h*HS + d]  (the transpose(0,2,1,3).reshape)
#pragma unroll
  for (int dt = 0; dt < 8; ++dt) {
    int dcol = h * HS_ + dt * 16 + ln16;
#pragma unroll
    for (int r = 0; r < 8; ++r) {
      int c = row0 + r + half8;
      ret[((size_t)bb * C_ + c) * E_ + dcol] = acc[dt][r];
    }
  }
}

// --------- state update: next_h = (k*last)^T @ v + hstate*kappa^512 ----------
// kbT already carries the kappa^(C-1-c) factor (folded in k_proj mode 1).
__global__ __launch_bounds__(256) void k_state(const __bf16* __restrict__ kbT,
                                               const __bf16* __restrict__ vbT,
                                               const float* __restrict__ hstate,
                                               float* __restrict__ nh) {
  int wv = threadIdx.x >> 5, ln = threadIdx.x & 31;
  int bh = blockIdx.x;
  int h  = bh & (H_ - 1);
  const __bf16* k_ = kbT + (size_t)bh * HS_ * C_;
  const __bf16* v_ = vbT + (size_t)bh * HS_ * C_;
  float cd = __expf(lkappa(h) * (float)(C_ / 4));   // chunk_decay = kappa^512
  int m0 = wv * 16;
  int ln16 = ln & 15, half8 = (ln & 16) ? 8 : 0;
  v8f acc[8];
#pragma unroll
  for (int t = 0; t < 8; ++t) {
    int n = t * 16 + ln16;
#pragma unroll
    for (int r = 0; r < 8; ++r) {
      int m = m0 + r + half8;
      acc[t][r] = hstate[((size_t)bh * HS_ + m) * HS_ + n] * cd;
    }
  }
  for (int s = 0; s < C_ / 32; ++s) {
    if (s + 1 < C_ / 32)
      __builtin_prefetch(k_ + (size_t)(m0 + (ln & 15)) * C_ + (s + 1) * 32, 0, 1);
    v16bf a = load_a_bf16(k_, C_, m0, s * 32, ln);
#pragma unroll
    for (int t = 0; t < 8; ++t) {
      v16bf b = load_b_bf16(v_, C_, t * 16, s * 32, ln);
      acc[t] = wmma_bf16(a, b, acc[t]);
    }
  }
#pragma unroll
  for (int t = 0; t < 8; ++t) {
    int n = t * 16 + ln16;
#pragma unroll
    for (int r = 0; r < 8; ++r) {
      int m = m0 + r + half8;
      nh[((size_t)bh * HS_ + m) * HS_ + n] = acc[t][r];
    }
  }
}

// ---------------- gate GEMM + GroupNorm (register-resident) ----------------
__global__ __launch_bounds__(128) void k_gate_gn(const __bf16* __restrict__ bkey,
                                                 const __bf16* __restrict__ wgT,
                                                 const float* __restrict__ ret,
                                                 const float* __restrict__ gamma,
                                                 const float* __restrict__ beta,
                                                 __bf16* __restrict__ gout) {
  int wv = threadIdx.x >> 5, ln = threadIdx.x & 31;
  int row0 = blockIdx.x * 64 + wv * 16;
  int hh = blockIdx.y;              // 128-col tile == group (head)
  int n0 = hh * HS_;
  v8f z = {}; v8f ag[8];
#pragma unroll
  for (int t = 0; t < 8; ++t) ag[t] = z;
  for (int ks = 0; ks < E_ / 32; ++ks) {
    if (ks + 1 < E_ / 32)
      __builtin_prefetch(bkey + (size_t)(row0 + (ln & 15)) * E_ + (ks + 1) * 32, 0, 1);
    v16bf a = load_a_bf16(bkey, E_, row0, ks * 32, ln);
#pragma unroll
    for (int t = 0; t < 8; ++t) {
      v16bf b = load_b_bf16(wgT, E_, n0 + t * 16, ks * 32, ln);
      ag[t] = wmma_bf16(a, b, ag[t]);
    }
  }
  int ln16 = ln & 15, half8 = (ln & 16) ? 8 : 0;

  float rr[64];
#pragma unroll
  for (int t = 0; t < 8; ++t) {
    int col = n0 + t * 16 + ln16;
#pragma unroll
    for (int r = 0; r < 8; ++r) {
      int rg = row0 + r + half8;
      rr[t * 8 + r] = ret[(size_t)rg * E_ + col];
    }
  }
  // per-row mean/var: 128 values of a row live in one 16-lane half (8 per lane)
  float mean[8], rstd[8];
#pragma unroll
  for (int r = 0; r < 8; ++r) {
    float sm = 0.f, sq = 0.f;
#pragma unroll
    for (int t = 0; t < 8; ++t) { float v = rr[t * 8 + r]; sm += v; sq += v * v; }
#pragma unroll
    for (int m = 1; m < 16; m <<= 1) {
      sm += __shfl_xor(sm, m, 32);
      sq += __shfl_xor(sq, m, 32);
    }
    float mu  = sm * (1.0f / HS_);
    float var = sq * (1.0f / HS_) - mu * mu;
    mean[r] = mu;
    rstd[r] = rsqrtf(var + 1e-5f);
  }
#pragma unroll
  for (int t = 0; t < 8; ++t) {
    int col = n0 + t * 16 + ln16;
    float gm = gamma[col], bt = beta[col];
#pragma unroll
    for (int r = 0; r < 8; ++r) {
      int rg = row0 + r + half8;
      float g = ag[t][r];
      float gate = g / (1.0f + __expf(-g));        // silu
      float gn = (rr[t * 8 + r] - mean[r]) * rstd[r] * gm + bt;
      gout[(size_t)rg * E_ + col] = f2bf(gate * gn);
    }
  }
}

// ---------------- final GEMM: out = gGN @ w_o ----------------
__global__ __launch_bounds__(128) void k_out(const __bf16* __restrict__ gin,
                                             const __bf16* __restrict__ woT,
                                             float* __restrict__ out) {
  int wv = threadIdx.x >> 5, ln = threadIdx.x & 31;
  int row0 = blockIdx.x * 64 + wv * 16;
  int n0 = blockIdx.y * 128;
  v8f z = {}; v8f acc[8];
#pragma unroll
  for (int t = 0; t < 8; ++t) acc[t] = z;
  for (int ks = 0; ks < E_ / 32; ++ks) {
    if (ks + 1 < E_ / 32)
      __builtin_prefetch(gin + (size_t)(row0 + (ln & 15)) * E_ + (ks + 1) * 32, 0, 1);
    v16bf a = load_a_bf16(gin, E_, row0, ks * 32, ln);
#pragma unroll
    for (int t = 0; t < 8; ++t) {
      v16bf b = load_b_bf16(woT, E_, n0 + t * 16, ks * 32, ln);
      acc[t] = wmma_bf16(a, b, acc[t]);
    }
  }
  int ln16 = ln & 15, half8 = (ln & 16) ? 8 : 0;
#pragma unroll
  for (int t = 0; t < 8; ++t) {
    int col = n0 + t * 16 + ln16;
#pragma unroll
    for (int r = 0; r < 8; ++r) {
      int rg = row0 + r + half8;
      out[(size_t)rg * E_ + col] = acc[t][r];
    }
  }
}

// ---------------- host launcher ----------------
extern "C" void kernel_launch(void* const* d_in, const int* in_sizes, int n_in,
                              void* d_out, int out_size, void* d_ws, size_t ws_size,
                              hipStream_t stream) {
  (void)in_sizes; (void)n_in; (void)out_size; (void)ws_size;
  const float* key    = (const float*)d_in[0];
  const float* query  = (const float*)d_in[1];
  const float* value  = (const float*)d_in[2];
  const float* hstate = (const float*)d_in[3];
  const float* w_q    = (const float*)d_in[4];
  const float* w_k    = (const float*)d_in[5];
  const float* w_v    = (const float*)d_in[6];
  const float* w_g    = (const float*)d_in[7];
  const float* w_o    = (const float*)d_in[8];
  const float* gamma  = (const float*)d_in[9];
  const float* beta   = (const float*)d_in[10];

  float* out    = (float*)d_out;
  float* next_h = out + (size_t)BC_ * E_;

  char* ws = (char*)d_ws;
  size_t off = 0;
  auto alloc = [&](size_t bytes) -> char* {
    char* p = ws + off;
    off += (bytes + 255) & ~(size_t)255;
    return p;
  };
  const size_t EE   = (size_t)E_ * E_;
  const size_t BCE  = (size_t)BC_ * E_;               // 8.39M elems
  __bf16* wqT  = (__bf16*)alloc(EE * 2);
  __bf16* wkT  = (__bf16*)alloc(EE * 2);
  __bf16* wvT  = (__bf16*)alloc(EE * 2);
  __bf16* wgT  = (__bf16*)alloc(EE * 2);
  __bf16* woT  = (__bf16*)alloc(EE * 2);
  __bf16* hT   = (__bf16*)alloc((size_t)B_ * H_ * HS_ * HS_ * 2);
  __bf16* bkey = (__bf16*)alloc(BCE * 2);
  __bf16* bqry = (__bf16*)alloc(BCE * 2);
  __bf16* bval = (__bf16*)alloc(BCE * 2);
  __bf16* qb   = (__bf16*)alloc(BCE * 2);
  __bf16* kb   = (__bf16*)alloc(BCE * 2);
  __bf16* kbT  = (__bf16*)alloc(BCE * 2);             // pre-scaled by last[c]
  __bf16* vbT  = (__bf16*)alloc(BCE * 2);
  float*  ret  = (float*)alloc(BCE * 4);
  __bf16* gGN  = (__bf16*)alloc(BCE * 2);

  // 1) one-shot bf16 conversion of activations (memory bound, done once)
  int n4 = (int)(BCE / 4);
  k_cvt_in<<<(n4 + 255) / 256, 256, 0, stream>>>(key,   bkey, n4);
  k_cvt_in<<<(n4 + 255) / 256, 256, 0, stream>>>(query, bqry, n4);
  k_cvt_in<<<(n4 + 255) / 256, 256, 0, stream>>>(value, bval, n4);

  // weights / state to bf16 (transposed for contiguous B-fragment loads)
  int nEE = E_ * E_;
  k_cvt_wqkv<<<(nEE + 255) / 256, 256, 0, stream>>>(w_q, wqT);
  k_cvt_wqkv<<<(nEE + 255) / 256, 256, 0, stream>>>(w_k, wkT);
  k_cvt_wqkv<<<(nEE + 255) / 256, 256, 0, stream>>>(w_v, wvT);
  k_cvt_sq  <<<(nEE + 255) / 256, 256, 0, stream>>>(w_g, wgT);
  k_cvt_sq  <<<(nEE + 255) / 256, 256, 0, stream>>>(w_o, woT);
  int nH = B_ * H_ * HS_ * HS_;
  k_cvt_h<<<(nH + 255) / 256, 256, 0, stream>>>(hstate, hT);

  // 2) projections
  dim3 gproj(BC_ / 64, E_ / 128);
  k_proj<<<gproj, 128, 0, stream>>>(bqry, wqT, qb, qb, 0);
  k_proj<<<gproj, 128, 0, stream>>>(bkey, wkT, kb, kbT, 1);
  k_proj<<<gproj, 128, 0, stream>>>(bval, wvT, vbT, vbT, 2);

  // 3) retention core -> ret (fp32, [B,C,E] layout)
  dim3 gattn(C_ / 64, H_, B_);
  k_attn<<<gattn, 128, 0, stream>>>(qb, kb, vbT, hT, ret);

  // 4) state update -> next_h
  k_state<<<B_ * H_, 256, 0, stream>>>(kbT, vbT, hstate, next_h);

  // 5) gate + groupnorm -> gGN (bf16)
  dim3 ggn(BC_ / 64, E_ / 128);
  k_gate_gn<<<ggn, 128, 0, stream>>>(bkey, wgT, ret, gamma, beta, gGN);

  // 6) output GEMM -> out
  dim3 gout(BC_ / 64, E_ / 128);
  k_out<<<gout, 128, 0, stream>>>(gGN, woT, out);
}
